// CosteffEncoder_53420803227876
// MI455X (gfx1250) — matrix operand order repaired and checked by reading, not compiled, not run on hardware
//
#include <hip/hip_runtime.h>
#include <hip/hip_bf16.h>
#include <math.h>
#include <stdint.h>

// ---------------- problem constants (from reference) ----------------
#define LYR 12
#define BB  8
#define SS  512
#define HH  768
#define NHD 12
#define HD  64
#define FF  3072
#define NL  2
#define MROWS (BB*SS)          // 4096 GEMM rows

typedef __bf16 bf16;
typedef bf16  v8bf  __attribute__((ext_vector_type(8)));
typedef bf16  v16bf __attribute__((ext_vector_type(16)));
typedef float v8f   __attribute__((ext_vector_type(8)));

static __device__ __forceinline__ v16bf cat16(v8bf lo, v8bf hi) {
  return __builtin_shufflevector(lo, hi, 0,1,2,3,4,5,6,7,8,9,10,11,12,13,14,15);
}
// A fragment (16x32 bf16, row-major source, row = lane&15):
//   lanes 0-15 : elems 0-15 -> K 0-15
//   lanes 16-31: elems 0-15 -> K 16-31
static __device__ __forceinline__ v16bf load_afrag(const bf16* rowBase, int kb, int hi) {
  v8bf lo = *(const v8bf*)(rowBase + kb + hi * 16);
  v8bf hiv = *(const v8bf*)(rowBase + kb + hi * 16 + 8);
  return cat16(lo, hiv);
}
// B fragment (32x16 bf16) from K-contiguous (pre-transposed) source, col = lane&15:
//   lanes 0-15: K 0..15 ; lanes 16-31: K 16..31
static __device__ __forceinline__ v16bf load_bfrag(const bf16* colBase, int kb, int hi) {
  v8bf lo = *(const v8bf*)(colBase + kb + hi * 16);
  v8bf hiv = *(const v8bf*)(colBase + kb + hi * 16 + 8);
  return cat16(lo, hiv);
}

static __device__ __forceinline__ float gelu_exact(float x) {
  return 0.5f * x * (1.0f + erff(x * 0.70710678118654752f));
}

// Async copy one 16B chunk global -> LDS (CDNA5 ASYNCcnt path).
static __device__ __forceinline__ void async_b128(const bf16* src, void* ldsDst) {
  uint32_t dst = (uint32_t)(uintptr_t)ldsDst;   // flat LDS aperture: low 32 = LDS addr
  uint64_t sa  = (uint64_t)(uintptr_t)src;
  asm volatile("global_load_async_to_lds_b128 %0, %1, off"
               :: "v"(dst), "v"(sa) : "memory");
}
static __device__ __forceinline__ void wait_async0() {
  asm volatile("s_wait_asynccnt 0x0" ::: "memory");
}

// Scheduler pipelining hints (group masks: 0x008 WMMA, 0x020 VMEM read, 0x100 DS read)
#define SCHED_DS_READS(n)   __builtin_amdgcn_sched_group_barrier(0x100, (n), 0)
#define SCHED_VMEM_READS(n) __builtin_amdgcn_sched_group_barrier(0x020, (n), 0)
#define SCHED_WMMAS(n)      __builtin_amdgcn_sched_group_barrier(0x008, (n), 0)

// LDS bank-conflict swizzle for the fp32 score slab (row stride 512 == 0 mod 64
// banks): XOR column bits 2..5 with the row id so 16 rows at one logical column
// land in 16 distinct banks.
static __device__ __forceinline__ int scidx(int row, int col) {
  return col ^ ((row & 15) << 2);
}

// ---------------- init: x -> h (f32) + hb (bf16) ----------------
__global__ void k_init(const float* __restrict__ x, float* __restrict__ h,
                       bf16* __restrict__ hb, int n) {
  int i = blockIdx.x * blockDim.x + threadIdx.x;
  if (i < n) { float v = x[i]; h[i] = v; hb[i] = (bf16)v; }
}

// ---------------- weight prep: fp32 [K,N] -> bf16 transposed [N,K] ----------------
__global__ void k_wprep(const float* __restrict__ src, bf16* __restrict__ dst,
                        int K, int N) {
  int i = blockIdx.x * blockDim.x + threadIdx.x;
  if (i < K * N) {
    int k = i / N, n = i - k * N;
    dst[(size_t)n * K + k] = (bf16)src[i];
  }
}

// ---------------- V transpose: v[B,S,H] bf16 -> vt[B,NH,HD,S] bf16 ----------------
__global__ void k_vtrans(const bf16* __restrict__ v, bf16* __restrict__ vt, int n) {
  int i = blockIdx.x * blockDim.x + threadIdx.x;
  if (i < n) {
    int b = i / (SS * HH);
    int r = i - b * (SS * HH);
    int s = r / HH;
    int hh = r - s * HH;
    int nh = hh / HD, hd = hh - nh * HD;
    vt[(((size_t)(b * NHD + nh)) * HD + hd) * SS + s] = v[i];
  }
}

// ---------------- WMMA GEMM: out = act(A[M,K]@Wt^T + bias (+resid)) ----------------
// A bf16 row-major [M,K]; Wt bf16 [N,K] (weight pre-transposed, K contiguous).
// 256 threads = 8 waves; wave -> 16Mx64N strip; block tile 128Mx64N.
// Weight tile (64N x 32K) double-buffered in LDS via async global->LDS copies.
// Sched hints pack 8 ds_load_b128 into one clause -> one DS wait -> 4
// back-to-back WMMAs; next A fragment prefetched from global in parallel.
__global__ void k_gemm(const bf16* __restrict__ A, const bf16* __restrict__ Wt,
                       const float* __restrict__ bias, const float* __restrict__ resid,
                       float* __restrict__ outF, bf16* __restrict__ outB,
                       int M, int K, int N, int act) {
  __shared__ bf16 Bs[2][64][32];      // 8 KB double-buffered weight tile
  const int tid  = threadIdx.x;
  const int lane = tid & 31;
  const int w    = tid >> 5;
  const int hi   = lane >> 4;
  const int lo   = lane & 15;
  const int mBase = blockIdx.x * 128 + w * 16;
  const int nBase = blockIdx.y * 64;

  // cooperative async B-tile fetch: thread t -> row n = t/4, 8-elem chunk t%4
  const int cpN  = tid >> 2;
  const int cpC  = (tid & 3) * 8;
  const bf16* cpSrcRow = Wt + (size_t)(nBase + cpN) * K + cpC;

  const v8f z = {0.f,0.f,0.f,0.f,0.f,0.f,0.f,0.f};
  v8f acc[4] = {z, z, z, z};

  const bf16* aRow = A + (size_t)(mBase + lo) * K;

  const int nsteps = K / 32;
  async_b128(cpSrcRow, &Bs[0][cpN][cpC]);            // prime buffer 0
  v16bf a_cur = load_afrag(aRow, 0, hi);

  for (int i = 0; i < nsteps; ++i) {
    wait_async0();
    __syncthreads();                                  // tile i resident for all waves
    if (i + 1 < nsteps)                               // stream tile i+1 into other buf
      async_b128(cpSrcRow + (i + 1) * 32, &Bs[(i + 1) & 1][cpN][cpC]);
    v16bf a_next = (i + 1 < nsteps) ? load_afrag(aRow, (i + 1) * 32, hi) : a_cur;

    const int buf = i & 1;
    v16bf bfr[4];
#pragma unroll
    for (int t = 0; t < 4; ++t) {
      v8bf blo = *(const v8bf*)&Bs[buf][t * 16 + lo][hi * 16];
      v8bf bhi = *(const v8bf*)&Bs[buf][t * 16 + lo][hi * 16 + 8];
      bfr[t] = cat16(blo, bhi);
    }
#pragma unroll
    for (int t = 0; t < 4; ++t)
      acc[t] = __builtin_amdgcn_wmma_f32_16x16x32_bf16(
          false, a_cur, false, bfr[t], (short)0, acc[t], false, false);
    // schedule: all 8 DS reads first (one wait), then 4 WMMAs back-to-back
    SCHED_DS_READS(8);
    SCHED_WMMAS(4);
    a_cur = a_next;
  }

  // Epilogue. C layout: lane holds col (lane&15); vgpr v -> row hi*8+v.
#pragma unroll
  for (int t = 0; t < 4; ++t) {
    int c = nBase + t * 16 + lo;
    float bia = bias[c];
#pragma unroll
    for (int v = 0; v < 8; ++v) {
      int r = mBase + hi * 8 + v;
      float x = acc[t][v] + bia;
      if (resid) x += resid[(size_t)r * N + c];
      if (act == 1) x = gelu_exact(x);
      if (outF) outF[(size_t)r * N + c] = x;
      if (outB) outB[(size_t)r * N + c] = (bf16)x;
    }
  }
}

// ---------------- fused attention ----------------
// grid: (S/32, NH, B); 64 threads = 2 waves; each wave owns a 16-query tile.
// q,k: bf16 [B,S,H]; vt: bf16 [B,NH,HD,S]; mask: f32 [B,S]; ctx: bf16 [B,S,H]
__global__ void k_attn(const bf16* __restrict__ q, const bf16* __restrict__ kmat,
                       const bf16* __restrict__ vt, const float* __restrict__ mask,
                       bf16* __restrict__ ctx) {
  __shared__ float sc[2][16][SS];   // 64 KB: per-wave 16x512 fp32 score slab
  const int lane = threadIdx.x & 31;
  const int w    = threadIdx.x >> 5;
  const int hi   = lane >> 4;
  const int lo   = lane & 15;
  const int b = blockIdx.z, h = blockIdx.y;
  const int qBase = blockIdx.x * 32 + w * 16;

  const bf16* qRow = q + ((size_t)(b * SS + qBase + lo)) * HH + h * HD;
  const v8f z = {0.f,0.f,0.f,0.f,0.f,0.f,0.f,0.f};

  // Q fragments are invariant across key tiles: load once.
  v16bf aq0 = load_afrag(qRow, 0, hi);
  v16bf aq1 = load_afrag(qRow, 32, hi);

  // ---- phase 1: S = Q K^T * scale + mask  -> LDS (swizzled)
  for (int kt = 0; kt < SS / 16; ++kt) {
    const bf16* kRow = kmat + ((size_t)(b * SS + kt * 16 + lo)) * HH + h * HD;
    v16bf b0 = load_bfrag(kRow, 0, hi);
    v16bf b1 = load_bfrag(kRow, 32, hi);
    v8f s = z;
    s = __builtin_amdgcn_wmma_f32_16x16x32_bf16(false, aq0, false, b0, (short)0, s, false, false);
    s = __builtin_amdgcn_wmma_f32_16x16x32_bf16(false, aq1, false, b1, (short)0, s, false, false);
    SCHED_VMEM_READS(4);
    SCHED_WMMAS(2);
    float mv = mask[b * SS + kt * 16 + lo];
#pragma unroll
    for (int v = 0; v < 8; ++v) {
      int r = hi * 8 + v;
      sc[w][r][scidx(r, kt * 16 + lo)] = s[v] * 0.125f + mv;
    }
  }

  // ---- phase 2: in-place fp32 softmax (2 lanes per row, swizzled access)
  {
    int rrow = lane >> 1, half = lane & 1;
    float* srow = sc[w][rrow];
    float m = -3.4e38f;
    for (int j = 0; j < SS / 2; ++j) m = fmaxf(m, srow[scidx(rrow, 2 * j + half)]);
    m = fmaxf(m, __shfl_xor(m, 1, 32));
    float l = 0.f;
    for (int j = 0; j < SS / 2; ++j) {
      int idx = scidx(rrow, 2 * j + half);
      float e = expf(srow[idx] - m);
      srow[idx] = e; l += e;
    }
    l += __shfl_xor(l, 1, 32);
    float inv = 1.f / l;
    for (int j = 0; j < SS / 2; ++j) srow[scidx(rrow, 2 * j + half)] *= inv;
  }

  // ---- phase 3: ctx = P V  (P from LDS fp32 -> bf16 A-frags; V^T contiguous)
  v8f o[4] = {z, z, z, z};
  const float* arow = sc[w][lo];
  const bf16* vBase = vt + (((size_t)(b * NHD + h)) * HD + lo) * SS;
  for (int kb = 0; kb < SS; kb += 32) {
    v16bf a;
#pragma unroll
    for (int e = 0; e < 8; ++e) a[e]     = (bf16)arow[scidx(lo, kb + hi * 16 + e)];
#pragma unroll
    for (int e = 0; e < 8; ++e) a[8 + e] = (bf16)arow[scidx(lo, kb + hi * 16 + 8 + e)];
    v16bf bfr[4];
#pragma unroll
    for (int t = 0; t < 4; ++t)
      bfr[t] = load_bfrag(vBase + (size_t)(t * 16) * SS, kb, hi);
#pragma unroll
    for (int t = 0; t < 4; ++t)
      o[t] = __builtin_amdgcn_wmma_f32_16x16x32_bf16(
          false, a, false, bfr[t], (short)0, o[t], false, false);
    // schedule: prob DS reads + V^T VMEM reads first, then 4 WMMAs
    SCHED_DS_READS(16);
    SCHED_VMEM_READS(8);
    SCHED_WMMAS(4);
  }
#pragma unroll
  for (int t = 0; t < 4; ++t)
#pragma unroll
    for (int v = 0; v < 8; ++v)
      ctx[((size_t)(b * SS + qBase + hi * 8 + v)) * HH + h * HD + t * 16 + lo] =
          (bf16)o[t][v];
}

// ---------------- LayerNorm (block per row over H=768) ----------------
__global__ void k_ln(const float* __restrict__ x, const float* __restrict__ g,
                     const float* __restrict__ b, float* __restrict__ outF,
                     bf16* __restrict__ outB) {
  __shared__ float rs[256], rq[256];
  const int row = blockIdx.x, tid = threadIdx.x;
  const float* xr = x + (size_t)row * HH;
  float s = 0.f, qq = 0.f;
  for (int j = tid; j < HH; j += 256) { float v = xr[j]; s += v; qq += v * v; }
  rs[tid] = s; rq[tid] = qq; __syncthreads();
  for (int off = 128; off > 0; off >>= 1) {
    if (tid < off) { rs[tid] += rs[tid + off]; rq[tid] += rq[tid + off]; }
    __syncthreads();
  }
  float mean = rs[0] * (1.0f / HH);
  float var  = rq[0] * (1.0f / HH) - mean * mean;
  float rstd = rsqrtf(var + 1e-12f);
  for (int j = tid; j < HH; j += 256) {
    float v = (xr[j] - mean) * rstd * g[j] + b[j];
    outF[(size_t)row * HH + j] = v;
    if (outB) outB[(size_t)row * HH + j] = (bf16)v;
  }
}

// ---------------- pooler(tanh cls) -> classifier -> entropy ----------------
// 1 block x 512 threads: wave w (0..15) -> (b = w/2, c = w%2) 768-dot.
__global__ void k_cls(const float* __restrict__ hfin, const float* __restrict__ cw,
                      const float* __restrict__ cb, float* __restrict__ logitsOut,
                      float* __restrict__ entOut) {
  __shared__ float lg[BB][NL];
  const int tid = threadIdx.x, lane = tid & 31, w = tid >> 5;
  if (w < BB * NL) {
    int b = w >> 1, c = w & 1;
    const float* hr = hfin + (size_t)(b * SS) * HH;  // cls token row
    float s = 0.f;
    for (int j = lane; j < HH; j += 32) s += tanhf(hr[j]) * cw[j * NL + c];
#pragma unroll
    for (int off = 16; off > 0; off >>= 1) s += __shfl_xor(s, off, 32);
    if (lane == 0) {
      float lv = s + cb[c];
      lg[b][c] = lv;
      logitsOut[b * NL + c] = lv;
    }
  }
  __syncthreads();
  if (tid < BB) {
    float l0 = lg[tid][0], l1 = lg[tid][1];
    float m = fmaxf(l0, l1);
    float z0 = expf(l0 - m), z1 = expf(l1 - m);
    float Z = z0 + z1, lse = m + logf(Z);
    float p0 = z0 / Z, p1 = z1 / Z;
    entOut[tid] = -(p0 * (l0 - lse) + p1 * (l1 - lse));
  }
}

// =====================================================================
extern "C" void kernel_launch(void* const* d_in, const int* in_sizes, int n_in,
                              void* d_out, int out_size, void* d_ws, size_t ws_size,
                              hipStream_t stream) {
  (void)in_sizes; (void)n_in; (void)out_size; (void)ws_size;
  const float* x    = (const float*)d_in[0];
  const float* mask = (const float*)d_in[1];
  const float* Wq = (const float*)d_in[2];  const float* bq = (const float*)d_in[3];
  const float* Wk = (const float*)d_in[4];  const float* bk = (const float*)d_in[5];
  const float* Wv = (const float*)d_in[6];  const float* bv = (const float*)d_in[7];
  const float* Wo = (const float*)d_in[8];  const float* bo = (const float*)d_in[9];
  const float* g1 = (const float*)d_in[10]; const float* b1 = (const float*)d_in[11];
  const float* Wi = (const float*)d_in[12]; const float* bi = (const float*)d_in[13];
  const float* Wd = (const float*)d_in[14]; const float* bd = (const float*)d_in[15];
  const float* g2 = (const float*)d_in[16]; const float* b2 = (const float*)d_in[17];
  const float* cw = (const float*)d_in[18]; const float* cb = (const float*)d_in[19];

  float* outFinal  = (float*)d_out;                          // B*S*H
  float* outLogits = outFinal + (size_t)BB * SS * HH;        // L*B*NL
  float* outEnts   = outLogits + (size_t)LYR * BB * NL;      // L*B

  // ---- carve workspace
  char* p = (char*)d_ws;
  auto carve = [&](size_t bytes) -> char* {
    char* r = p; p += (bytes + 255) & ~(size_t)255; return r;
  };
  const size_t nBSH = (size_t)BB * SS * HH;
  float* h     = (float*)carve(nBSH * 4);
  bf16*  hb    = (bf16*) carve(nBSH * 2);
  bf16*  qb    = (bf16*) carve(nBSH * 2);
  bf16*  kb_   = (bf16*) carve(nBSH * 2);
  bf16*  vb    = (bf16*) carve(nBSH * 2);
  bf16*  vt    = (bf16*) carve(nBSH * 2);
  bf16*  ctx   = (bf16*) carve(nBSH * 2);
  float* pre   = (float*)carve(nBSH * 4);
  float* attnF = (float*)carve(nBSH * 4);
  bf16*  attnB = (bf16*) carve(nBSH * 2);
  bf16*  inter = (bf16*) carve((size_t)BB * SS * FF * 2);
  bf16*  WqT   = (bf16*) carve((size_t)HH * HH * 2);
  bf16*  WkT   = (bf16*) carve((size_t)HH * HH * 2);
  bf16*  WvT   = (bf16*) carve((size_t)HH * HH * 2);
  bf16*  WoT   = (bf16*) carve((size_t)HH * HH * 2);
  bf16*  WiT   = (bf16*) carve((size_t)HH * FF * 2);
  bf16*  WdT   = (bf16*) carve((size_t)FF * HH * 2);

  // ---- init h / hb from x
  {
    int n = (int)nBSH;
    k_init<<<dim3((n + 255) / 256), dim3(256), 0, stream>>>(x, h, hb, n);
  }

  const dim3 gHH(MROWS / 128, HH / 64);   // 32 x 12
  const dim3 gFF(MROWS / 128, FF / 64);   // 32 x 48
  const dim3 blk256(256);

  for (int l = 0; l < LYR; ++l) {
    const size_t oHH = (size_t)l * HH * HH;
    const size_t oHF = (size_t)l * HH * FF;
    // weight prep (fp32 -> transposed bf16)
    {
      int nhh = HH * HH, nhf = HH * FF;
      k_wprep<<<dim3((nhh + 255) / 256), blk256, 0, stream>>>(Wq + oHH, WqT, HH, HH);
      k_wprep<<<dim3((nhh + 255) / 256), blk256, 0, stream>>>(Wk + oHH, WkT, HH, HH);
      k_wprep<<<dim3((nhh + 255) / 256), blk256, 0, stream>>>(Wv + oHH, WvT, HH, HH);
      k_wprep<<<dim3((nhh + 255) / 256), blk256, 0, stream>>>(Wo + oHH, WoT, HH, HH);
      k_wprep<<<dim3((nhf + 255) / 256), blk256, 0, stream>>>(Wi + oHF, WiT, HH, FF);
      k_wprep<<<dim3((nhf + 255) / 256), blk256, 0, stream>>>(Wd + oHF, WdT, FF, HH);
    }
    // Q,K,V projections (bf16 out)
    k_gemm<<<gHH, blk256, 0, stream>>>(hb, WqT, bq + (size_t)l * HH, nullptr,
                                       nullptr, qb, MROWS, HH, HH, 0);
    k_gemm<<<gHH, blk256, 0, stream>>>(hb, WkT, bk + (size_t)l * HH, nullptr,
                                       nullptr, kb_, MROWS, HH, HH, 0);
    k_gemm<<<gHH, blk256, 0, stream>>>(hb, WvT, bv + (size_t)l * HH, nullptr,
                                       nullptr, vb, MROWS, HH, HH, 0);
    // V transpose for P@V B-fragments
    k_vtrans<<<dim3(((int)nBSH + 255) / 256), blk256, 0, stream>>>(vb, vt, (int)nBSH);
    // fused attention
    k_attn<<<dim3(SS / 32, NHD, BB), dim3(64), 0, stream>>>(qb, kb_, vt, mask, ctx);
    // output projection + residual(h) -> pre ; LN1 -> attnF/attnB
    k_gemm<<<gHH, blk256, 0, stream>>>(ctx, WoT, bo + (size_t)l * HH, h,
                                       pre, nullptr, MROWS, HH, HH, 0);
    k_ln<<<dim3(MROWS), blk256, 0, stream>>>(pre, g1 + (size_t)l * HH,
                                             b1 + (size_t)l * HH, attnF, attnB);
    // FFN up + GELU
    k_gemm<<<gFF, blk256, 0, stream>>>(attnB, WiT, bi + (size_t)l * FF, nullptr,
                                       nullptr, inter, MROWS, HH, FF, 1);
    // FFN down + residual(attnF) -> pre ; LN2 -> h/hb (last layer -> d_out)
    k_gemm<<<gHH, blk256, 0, stream>>>(inter, WdT, bd + (size_t)l * HH, attnF,
                                       pre, nullptr, MROWS, FF, HH, 0);
    float* hOut = (l == LYR - 1) ? outFinal : h;
    k_ln<<<dim3(MROWS), blk256, 0, stream>>>(pre, g2 + (size_t)l * HH,
                                             b2 + (size_t)l * HH, hOut, hb);
    // highway head
    k_cls<<<dim3(1), dim3(512), 0, stream>>>(hOut, cw + (size_t)l * HH * NL,
                                             cb + (size_t)l * NL,
                                             outLogits + (size_t)l * BB * NL,
                                             outEnts + (size_t)l * BB);
  }
}
